// LocalAttention_10943576670235
// MI455X (gfx1250) — matrix operand order repaired and checked by reading, not compiled, not run on hardware
//
#include <hip/hip_runtime.h>

typedef _Float16 v16h __attribute__((ext_vector_type(16)));
typedef _Float16 v8h  __attribute__((ext_vector_type(8)));
typedef _Float16 v4h  __attribute__((ext_vector_type(4)));
typedef float    v8f  __attribute__((ext_vector_type(8)));
typedef float    v4f  __attribute__((ext_vector_type(4)));

// Problem constants (match reference)
#define BN 2
#define CN 128
#define HN 64
#define WN 64
#define RAD 4          // 9x9 window radius
#define NDY 9
#define TW  16         // pixel tile width (M dim)
#define NX  32         // x-band columns per tile (two 16-wide N tiles)
#define SCALE_F 0.08838834764831843f  // 128^-0.5

// LDS pitches (in halves) — chosen so fragment base addresses are 16B aligned
#define QP 136         // 128 + 8
#define KP 136
#define VP 40          // 32 + 8

// LDS region byte offsets (all multiples of 16)
#define QS_OFF   0
#define QS_BYTES (TW * QP * 2)                       // 4352
#define KS_OFF   (QS_OFF + QS_BYTES)                 // 4352
#define KS_BYTES (NDY * NX * KP * 2)                 // 78336
#define VS_OFF   (KS_OFF + KS_BYTES)                 // 82688
#define VS_BYTES (NDY * CN * VP * 2)                 // 92160
#define SF_OFF   (VS_OFF + VS_BYTES)                 // 174848
#define SF_BYTES (NDY * NX * TW * 4)                 // 18432  ([dy][lx][m], m contiguous)
#define WH_OFF   (SF_OFF + SF_BYTES)                 // 193280
#define WH_BYTES (TW * NDY * NX * 2)                 // 9216   ([m][dy][lx], lx contiguous)
#define SMEM_TOTAL (WH_OFF + WH_BYTES)               // 202496 bytes

__launch_bounds__(256)
__global__ void local_attn_wmma_kernel(const float* __restrict__ gq,
                                       const float* __restrict__ gk,
                                       const float* __restrict__ gv,
                                       float* __restrict__ gout) {
    extern __shared__ char smem[];
    _Float16* Qs = (_Float16*)(smem + QS_OFF);  // [TW][QP]
    _Float16* Ks = (_Float16*)(smem + KS_OFF);  // [NDY][NX][KP]  (c contiguous)
    _Float16* Vs = (_Float16*)(smem + VS_OFF);  // [NDY][CN][VP]  (x contiguous)
    float*    Sf = (float*)   (smem + SF_OFF);  // [NDY][NX][TW]  scores f32, m contiguous
    _Float16* Wh = (_Float16*)(smem + WH_OFF);  // [TW][NDY][NX]  weights f16

    const int tid  = threadIdx.x;
    const int lane = tid & 31;
    const int wave = __builtin_amdgcn_readfirstlane(tid >> 5);  // scalar: keeps WMMA loops unmasked

    const int blk = blockIdx.x;
    const int wt = blk & 3;           // 4 tiles across W
    const int h  = (blk >> 2) & 63;
    const int b  = blk >> 8;
    const int w0 = wt * TW;
    const int xbase = w0 - 8;

    // ---------------- Stage: global -> LDS with f32 -> f16 conversion ----------------
    // Q tile: Qs[m][c]; float4 over 4 consecutive w positions (w0 mult of 16 -> aligned)
    for (int i = tid; i < TW * CN / 4; i += 256) {
        const int mg = (i & 3) * 4;
        const int c  = i >> 2;
        const v4f val = *(const v4f*)&gq[((b * CN + c) * HN + h) * WN + (w0 + mg)];
#pragma unroll
        for (int j = 0; j < 4; ++j)
            Qs[(mg + j) * QP + c] = (_Float16)val[j];
    }
    // K band: Ks[dy][lx][c]. xbase ≡ 0 (mod 8), so 4-wide x groups are fully in or
    // fully out of the image (SAME zero-padding) and 16B-aligned in global memory.
    for (int i = tid; i < NDY * CN * (NX / 4); i += 256) {
        const int lx = (i & 7) * 4;
        const int c  = (i >> 3) & 127;
        const int dy = i >> 10;
        const int hy = h + dy - RAD;
        const int x  = xbase + lx;
        v4f val = {};
        if ((unsigned)hy < (unsigned)HN && (unsigned)x < (unsigned)WN)
            val = *(const v4f*)&gk[((b * CN + c) * HN + hy) * WN + x];
#pragma unroll
        for (int j = 0; j < 4; ++j)
            Ks[(dy * NX + lx + j) * KP + c] = (_Float16)val[j];
    }
    // V band: Vs[dy][c][lx]; contiguous in lx -> one 8B LDS store per float4
    for (int i = tid; i < NDY * CN * (NX / 4); i += 256) {
        const int lx = (i & 7) * 4;
        const int c  = (i >> 3) & 127;
        const int dy = i >> 10;
        const int hy = h + dy - RAD;
        const int x  = xbase + lx;
        v4f val = {};
        if ((unsigned)hy < (unsigned)HN && (unsigned)x < (unsigned)WN)
            val = *(const v4f*)&gv[((b * CN + c) * HN + hy) * WN + x];
        v4h hv;
#pragma unroll
        for (int j = 0; j < 4; ++j)
            hv[j] = (_Float16)val[j];
        *(v4h*)&Vs[(dy * CN + c) * VP + lx] = hv;
    }
    __syncthreads();

    // Fragment lane coordinates (CDNA5 WMMA 16x16x32 f16 layouts)
    const int mrow  = lane & 15;               // A: M row / B,D: N column
    const int koffA = (lane < 16) ? 0 : 8;     // A: halves 0-7 -> K koffA..+7, 8-15 -> koffA+16..+23
    const int koffB = (lane < 16) ? 0 : 16;    // B: unused; B loads use koffA, same k order as A
    const int mD    = (lane < 16) ? 0 : 8;     // D: VGPR r -> M = r + mD

    // ---------------- Phase 1: scores S[m][lx] = sum_c Q[m][c] * K[c][lx], per dy -----
    // 18 output tiles (9 dy x 2 x-halves), round-robin over 8 waves, 4 WMMA (K=128) each.
    for (int t = wave; t < NDY * 2; t += 8) {
        const int dy  = t >> 1;
        const int lx0 = (t & 1) * 16;
        v8f acc = {};
#pragma unroll
        for (int kc = 0; kc < CN; kc += 32) {
            const _Float16* ap = &Qs[mrow * QP + kc + koffA];
            const v8h alo = *(const v8h*)ap;
            const v8h ahi = *(const v8h*)(ap + 16);
            const _Float16* bp = &Ks[(dy * NX + lx0 + mrow) * KP + kc + koffA];
            const v8h blo = *(const v8h*)bp;
            const v8h bhi = *(const v8h*)(bp + 16);
            v16h a, bm;
#pragma unroll
            for (int e = 0; e < 8; ++e) {
                a[e] = alo[e]; a[e + 8] = ahi[e];
                bm[e] = blo[e]; bm[e + 8] = bhi[e];
            }
            acc = __builtin_amdgcn_wmma_f32_16x16x32_f16(
                false, a, false, bm, (short)0, acc, false, false);
        }
        // D column (fixed N = lx0+mrow) is contiguous in m -> two b128 stores
        float* sp = &Sf[(dy * NX + lx0 + mrow) * TW + mD];
        const v4f s0 = {acc[0], acc[1], acc[2], acc[3]};
        const v4f s1 = {acc[4], acc[5], acc[6], acc[7]};
        *(v4f*)sp = s0;
        *(v4f*)(sp + 4) = s1;
    }
    __syncthreads();

    // ---------------- Softmax over the 81 in-window entries per pixel row ------------
    // Pixel m uses lx in [m+4, m+12] for each dy. Out-of-image entries have score 0
    // (K was zero-padded), matching the reference's zero-padded patches.
    {
        const int msm = tid >> 4;       // pixel row; lanes 0-15 / 16-31 split per wave
        const int sub = tid & 15;
        float mx = -1e30f;
        for (int j = sub; j < 81; j += 16) {
            const int dy = j / 9, dx = j % 9;
            mx = fmaxf(mx, Sf[(dy * NX + (msm + 4 + dx)) * TW + msm]);
        }
#pragma unroll
        for (int off = 8; off >= 1; off >>= 1)
            mx = fmaxf(mx, __shfl_xor(mx, off));   // stays within 16-lane group
        float sum = 0.0f;
        for (int j = sub; j < 81; j += 16) {
            const int dy = j / 9, dx = j % 9;
            sum += __expf(Sf[(dy * NX + (msm + 4 + dx)) * TW + msm] - mx);
        }
#pragma unroll
        for (int off = 8; off >= 1; off >>= 1)
            sum += __shfl_xor(sum, off);
        const float inv = SCALE_F / sum;           // reference scales AFTER softmax
        for (int j = sub; j < NDY * NX; j += 16) {
            const int dy = j >> 5, lx = j & 31;
            float wv = 0.0f;
            if (lx >= msm + 4 && lx <= msm + 12)
                wv = __expf(Sf[(dy * NX + lx) * TW + msm] - mx) * inv;
            Wh[msm * (NDY * NX) + j] = (_Float16)wv;
        }
    }
    __syncthreads();

    // ---------------- Phase 2: out[m][c] = sum_dy sum_lx W[m][dy][lx] * V[dy][c][lx] --
    // Each wave owns one 16-channel N tile; 9 WMMAs (K=32 per dy).
    {
        const int cb = wave * 16;
        v8f acc = {};
#pragma unroll
        for (int dy = 0; dy < NDY; ++dy) {
            const _Float16* ap = &Wh[(mrow * NDY + dy) * NX + koffA];
            const v8h alo = *(const v8h*)ap;
            const v8h ahi = *(const v8h*)(ap + 16);
            const _Float16* bp = &Vs[(dy * CN + cb + mrow) * VP + koffA];
            const v8h blo = *(const v8h*)bp;
            const v8h bhi = *(const v8h*)(bp + 16);
            v16h a, bm;
#pragma unroll
            for (int e = 0; e < 8; ++e) {
                a[e] = alo[e]; a[e + 8] = ahi[e];
                bm[e] = blo[e]; bm[e + 8] = bhi[e];
            }
            acc = __builtin_amdgcn_wmma_f32_16x16x32_f16(
                false, a, false, bm, (short)0, acc, false, false);
        }
        // D column (fixed channel) is 8 consecutive w positions -> two b128 stores
        float* op = &gout[((b * CN + cb + mrow) * HN + h) * WN + (w0 + mD)];
        const v4f o0 = {acc[0], acc[1], acc[2], acc[3]};
        const v4f o1 = {acc[4], acc[5], acc[6], acc[7]};
        *(v4f*)op = o0;
        *(v4f*)(op + 4) = o1;
    }
}

extern "C" void kernel_launch(void* const* d_in, const int* in_sizes, int n_in,
                              void* d_out, int out_size, void* d_ws, size_t ws_size,
                              hipStream_t stream) {
    (void)in_sizes; (void)n_in; (void)out_size; (void)d_ws; (void)ws_size;
    const float* q = (const float*)d_in[0];
    const float* k = (const float*)d_in[1];
    const float* v = (const float*)d_in[2];
    float* out = (float*)d_out;

    // 512 blocks: 4 w-tiles * 64 rows * 2 batches; 256 threads = 8 wave32
    local_attn_wmma_kernel<<<dim3((WN / TW) * HN * BN), dim3(256),
                             SMEM_TOTAL, stream>>>(q, k, v, out);
}
